// RelPositionMultiHeadAttention_41884521070710
// MI455X (gfx1250) — matrix operand order repaired and checked by reading, not compiled, not run on hardware
//
#include <hip/hip_runtime.h>
#include <hip/hip_bf16.h>
#include <math.h>

#define DIM 1024
#define NH  16
#define DH  64
#define BB  4
#define TT  1024

typedef __bf16 bf16_t;
typedef __attribute__((ext_vector_type(16))) __bf16 v16bf;
typedef __attribute__((ext_vector_type(8)))  __bf16 v8bf;
typedef __attribute__((ext_vector_type(8)))  float  v8f;

// ---------------- WMMA helpers ----------------

static __device__ inline v8f wmma_bf16(v16bf a, v16bf b, v8f c) {
  return __builtin_amdgcn_wmma_f32_16x16x32_bf16(
      /*neg_a=*/false, a, /*neg_b=*/false, b,
      /*c_mod=*/(short)0, c, /*reuse_a=*/false, /*reuse_b=*/false);
}

static __device__ inline v16bf cat16(v8bf lo, v8bf hi) {
  return __builtin_shufflevector(lo, hi, 0,1,2,3,4,5,6,7,8,9,10,11,12,13,14,15);
}

// A fragment (16x32, 16-bit): lane<16 -> row M=lane, K in {0..7, 16..23};
// lane>=16 -> row M=lane-16, K in {8..15, 24..31}.
static __device__ inline v16bf load_a_frag(const bf16_t* base, int ld) {
  int lane = threadIdx.x & 31;
  int row  = lane & 15;
  int koff = (lane >> 4) << 3;               // 0 or 8
  const bf16_t* p = base + (size_t)row * ld + koff;
  return cat16(*(const v8bf*)(p), *(const v8bf*)(p + 16));
}

// A fragment with explicit (clamped) row base: row = clamp(row0 + (lane&15)).
static __device__ inline v16bf load_a_frag_rows(const bf16_t* base, int ld,
                                                int row0, int kofs) {
  int lane = threadIdx.x & 31;
  int row  = row0 + (lane & 15);
  row = row < 0 ? 0 : (row > TT - 1 ? TT - 1 : row);
  int koff = ((lane >> 4) << 3) + kofs;
  const bf16_t* p = base + (size_t)row * ld + koff;
  return cat16(*(const v8bf*)(p), *(const v8bf*)(p + 16));
}

// B fragment (32x16, 16-bit), stored transposed (row = N, col = K):
// lane holds column N = lane&15; K in {0..7, 16..23} (lane<16) or {8..15, 24..31}.
static __device__ inline v16bf load_b_frag(const bf16_t* base, int ld) {
  int lane  = threadIdx.x & 31;
  int n     = lane & 15;
  int kbase = (lane >> 4) << 3;              // 0 or 8
  const bf16_t* p = base + (size_t)n * ld + kbase;
  return cat16(*(const v8bf*)(p), *(const v8bf*)(p + 16));
}

// B fragment with clamped row indices (rows index into p's T dimension).
static __device__ inline v16bf load_b_frag_rows(const bf16_t* base, int ld,
                                                int row0, int kofs) {
  int lane = threadIdx.x & 31;
  int row  = row0 + (lane & 15);
  row = row < 0 ? 0 : (row > TT - 1 ? TT - 1 : row);
  int kbase = ((lane >> 4) << 3) + kofs;
  const bf16_t* p = base + (size_t)row * ld + kbase;
  return cat16(*(const v8bf*)(p), *(const v8bf*)(p + 16));
}

// ---------------- conversion ----------------

__global__ void cvt_f32_bf16(const float* __restrict__ src,
                             bf16_t* __restrict__ dst, int n) {
  int i = blockIdx.x * blockDim.x + threadIdx.x;
  if (i < n) dst[i] = (bf16_t)src[i];
}

// ---------------- projection GEMM (16Mx64N per wave, A-frag reuse) --------
// C[M,DIM] = A[M,DIM] * Bt[DIM,DIM]^T (+bias, mode-dependent epilogue)
// mode 0: qu = acc+bias+bu, qv = acc+bias+bv  -> [B,H,T,DH] bf16 (out0, out1)
// mode 1: k  = acc+bias                       -> [B,H,T,DH] bf16 (out0)
// mode 2: v  = acc+bias                       -> [B,H,DH,T] bf16 (out0, transposed)
// mode 3: p  = acc                            -> [H,T,DH]   bf16 (out0)
// mode 4: out = acc+bias                      -> [B*T,DIM]  f32  (outf)
__global__ void proj_kernel(const bf16_t* __restrict__ A,
                            const bf16_t* __restrict__ Bt,
                            const float* __restrict__ bias,
                            const float* __restrict__ bu,
                            const float* __restrict__ bv,
                            bf16_t* __restrict__ out0,
                            bf16_t* __restrict__ out1,
                            float* __restrict__ outf,
                            int mode) {
  int m0 = blockIdx.x << 4;
  int n0 = blockIdx.y << 6;                       // 64-wide N block per wave
  int lane = threadIdx.x & 31;
  int ln   = lane & 15;
  int half = lane >> 4;

  // per-lane base pointers, hoisted out of the k-loop
  const bf16_t* pa  = A  + (size_t)(m0 + ln) * DIM + (half << 3);
  const bf16_t* pb0 = Bt + (size_t)(n0 + ln) * DIM + (half << 3);
  const bf16_t* pb1 = pb0 + (size_t)16 * DIM;
  const bf16_t* pb2 = pb0 + (size_t)32 * DIM;
  const bf16_t* pb3 = pb0 + (size_t)48 * DIM;

  v8f acc[4] = {v8f{}, v8f{}, v8f{}, v8f{}};
#pragma unroll 2
  for (int k0 = 0; k0 < DIM; k0 += 32) {
    // WGP-scope speculative prefetch 8 k-steps (256 elements) ahead (global_prefetch_b8)
    __builtin_prefetch(pa + k0 + 256, 0, 3);
    // batch all fragment loads, then all WMMAs -> incremental s_wait_loadcnt
    v16bf a  = cat16(*(const v8bf*)(pa + k0),  *(const v8bf*)(pa + k0 + 16));
    v16bf b0 = cat16(*(const v8bf*)(pb0 + k0), *(const v8bf*)(pb0 + k0 + 16));
    v16bf b1 = cat16(*(const v8bf*)(pb1 + k0), *(const v8bf*)(pb1 + k0 + 16));
    v16bf b2 = cat16(*(const v8bf*)(pb2 + k0), *(const v8bf*)(pb2 + k0 + 16));
    v16bf b3 = cat16(*(const v8bf*)(pb3 + k0), *(const v8bf*)(pb3 + k0 + 16));
    acc[0] = wmma_bf16(a, b0, acc[0]);
    acc[1] = wmma_bf16(a, b1, acc[1]);
    acc[2] = wmma_bf16(a, b2, acc[2]);
    acc[3] = wmma_bf16(a, b3, acc[3]);
  }

#pragma unroll
  for (int t = 0; t < 4; ++t) {
    int e  = n0 + (t << 4) + ln;
    int h  = e >> 6;
    int dh = e & (DH - 1);
#pragma unroll
    for (int r = 0; r < 8; ++r) {
      int row = m0 + r + (half << 3);
      float v = acc[t][r];
      if (mode == 4) {
        outf[(size_t)row * DIM + e] = v + bias[e];
      } else if (mode == 3) {
        out0[((size_t)h * TT + row) * DH + dh] = (bf16_t)v;
      } else {
        int bi = row >> 10;
        int tt = row & (TT - 1);
        float vb = v + bias[e];
        size_t hdt = (((size_t)bi * NH + h) * TT + tt) * DH + dh;
        if (mode == 0) {
          out0[hdt] = (bf16_t)(vb + bu[e]);
          out1[hdt] = (bf16_t)(vb + bv[e]);
        } else if (mode == 1) {
          out0[hdt] = (bf16_t)vb;
        } else { // mode 2: transposed v
          out0[(((size_t)bi * NH + h) * DH + dh) * TT + tt] = (bf16_t)vb;
        }
      }
    }
  }
}

// ---------------- flash attention with exact relative shift ----------------
// grid = (T/16, H, B), 32 threads (1 wave). 16 query rows per wave.
// relative_shift(x)[q,j] =  pscore[q,   T-1+j-q]  if j <= q
//                           0                      if j == q+1
//                           pscore[q+1, j-q-2]     if j >= q+2
// where pscore[r,m] = q_v[r] . p[m]. For a 16x16 (q,j) tile both branches read
// column col = 15+ji-qi of a 16x32 band: P1 rows q0..q0+15 at m0A = T-16+j0-q0,
// P2 rows q0+1..q0+16 at m0C = j0-q0-17. Bands computed with WMMA -> LDS,
// then gathered per element.
__global__ void attn_kernel(const bf16_t* __restrict__ qu,
                            const bf16_t* __restrict__ qv,
                            const bf16_t* __restrict__ kk,
                            const bf16_t* __restrict__ vT,
                            const bf16_t* __restrict__ pp,
                            const unsigned char* __restrict__ mask,
                            bf16_t* __restrict__ ctx) {
  int q0 = blockIdx.x << 4;
  int h  = blockIdx.y;
  int b  = blockIdx.z;
  int lane = threadIdx.x & 31;
  int ln   = lane & 15;
  int half = lane >> 4;

  const bf16_t* qu_h = qu + ((size_t)(b * NH + h) * TT) * DH;
  const bf16_t* qv_h = qv + ((size_t)(b * NH + h) * TT) * DH;
  const bf16_t* k_h  = kk + ((size_t)(b * NH + h) * TT) * DH;
  const bf16_t* v_h  = vT + ((size_t)(b * NH + h) * DH) * TT;
  const bf16_t* p_h  = pp + ((size_t)h * TT) * DH;
  const unsigned char* mask_b = mask + (size_t)b * TT * TT;

  // Query fragments (K = DH = 64 -> two 32-wide k-steps)
  v16bf au[2], avn[2], avs[2];
#pragma unroll
  for (int ks = 0; ks < 2; ++ks) {
    au[ks]  = load_a_frag(qu_h + (size_t)q0 * DH + ks * 32, DH);
    avn[ks] = load_a_frag(qv_h + (size_t)q0 * DH + ks * 32, DH);
    avs[ks] = load_a_frag_rows(qv_h, DH, q0 + 1, ks * 32);  // rows q0+1..q0+16
  }

  v8f o[4] = {v8f{}, v8f{}, v8f{}, v8f{}};    // ctx accumulator: 16 x 64
  float mrow[8], lrow[8];
#pragma unroll
  for (int r = 0; r < 8; ++r) { mrow[r] = -1e30f; lrow[r] = 0.0f; }

  __shared__ float  P1s[16][32];
  __shared__ float  P2s[16][32];
  __shared__ bf16_t As[16][32];

  for (int jt = 0; jt < TT; jt += 32) {
    float sc[2][8];
    for (int s = 0; s < 2; ++s) {
      int j0 = jt + (s << 4);
      // content scores: (q+u) . k  (batch loads, then WMMAs)
      v16bf bk0 = load_b_frag(k_h + (size_t)j0 * DH, DH);
      v16bf bk1 = load_b_frag(k_h + (size_t)j0 * DH + 32, DH);
      v8f c = {};
      c = wmma_bf16(au[0], bk0, c);
      c = wmma_bf16(au[1], bk1, c);
      // position bands
      int m0A = TT - 16 + j0 - q0;
      int m0C = j0 - q0 - 17;
      v8f p1[2] = {v8f{}, v8f{}};
      v8f p2[2] = {v8f{}, v8f{}};
#pragma unroll
      for (int t = 0; t < 2; ++t) {
        v16bf b1l = load_b_frag_rows(p_h, DH, m0A + 16 * t, 0);
        v16bf b1h = load_b_frag_rows(p_h, DH, m0A + 16 * t, 32);
        v16bf b2l = load_b_frag_rows(p_h, DH, m0C + 16 * t, 0);
        v16bf b2h = load_b_frag_rows(p_h, DH, m0C + 16 * t, 32);
        p1[t] = wmma_bf16(avn[0], b1l, p1[t]);
        p1[t] = wmma_bf16(avn[1], b1h, p1[t]);
        p2[t] = wmma_bf16(avs[0], b2l, p2[t]);
        p2[t] = wmma_bf16(avs[1], b2h, p2[t]);
      }
      __syncthreads();
#pragma unroll
      for (int t = 0; t < 2; ++t)
#pragma unroll
        for (int r = 0; r < 8; ++r) {
          P1s[r + (half << 3)][(t << 4) + ln] = p1[t][r];
          P2s[r + (half << 3)][(t << 4) + ln] = p2[t][r];
        }
      __syncthreads();
#pragma unroll
      for (int r = 0; r < 8; ++r) {
        int qi = r + (half << 3);
        int q  = q0 + qi;
        int j  = j0 + ln;
        int dd = j - q;
        int col = 15 + ln - qi;                 // identical for both bands
        float posv = (dd == 1) ? 0.0f : (dd <= 0 ? P1s[qi][col] : P2s[qi][col]);
        float v = (c[r] + posv) * 0.125f;       // 1/sqrt(DH)
        if (!mask_b[(size_t)q * TT + j]) v = -1e9f;
        sc[s][r] = v;
      }
      __syncthreads();
    }

    // online softmax update over this 32-wide block
    float fac[8], pv0[8], pv1[8];
#pragma unroll
    for (int r = 0; r < 8; ++r) {
      float mx = fmaxf(sc[0][r], sc[1][r]);
#pragma unroll
      for (int d = 1; d < 16; d <<= 1) mx = fmaxf(mx, __shfl_xor(mx, d, 32));
      float mn = fmaxf(mrow[r], mx);
      fac[r] = __expf(mrow[r] - mn);
      pv0[r] = __expf(sc[0][r] - mn);
      pv1[r] = __expf(sc[1][r] - mn);
      float rs = pv0[r] + pv1[r];
#pragma unroll
      for (int d = 1; d < 16; d <<= 1) rs += __shfl_xor(rs, d, 32);
      lrow[r] = lrow[r] * fac[r] + rs;
      mrow[r] = mn;
    }

    // attn tile (16x32) -> LDS -> A fragment
#pragma unroll
    for (int r = 0; r < 8; ++r) {
      As[r + (half << 3)][ln]      = (bf16_t)pv0[r];
      As[r + (half << 3)][16 + ln] = (bf16_t)pv1[r];
    }
    __syncthreads();
    v16bf af = load_a_frag(&As[0][0], 32);

    // batch the 4 v fragments, then rescale + 4 WMMAs
    v16bf bv0 = load_b_frag(v_h + (size_t)0  * TT + jt, TT);
    v16bf bv1 = load_b_frag(v_h + (size_t)16 * TT + jt, TT);
    v16bf bv2 = load_b_frag(v_h + (size_t)32 * TT + jt, TT);
    v16bf bv3 = load_b_frag(v_h + (size_t)48 * TT + jt, TT);
#pragma unroll
    for (int t = 0; t < 4; ++t)
#pragma unroll
      for (int r = 0; r < 8; ++r) o[t][r] *= fac[r];
    o[0] = wmma_bf16(af, bv0, o[0]);
    o[1] = wmma_bf16(af, bv1, o[1]);
    o[2] = wmma_bf16(af, bv2, o[2]);
    o[3] = wmma_bf16(af, bv3, o[3]);
    __syncthreads();
  }

  // finalize: ctx[b, q, h*DH + d] = O / l
#pragma unroll
  for (int r = 0; r < 8; ++r) {
    int qi = r + (half << 3);
    float inv = 1.0f / lrow[r];
#pragma unroll
    for (int t = 0; t < 4; ++t) {
      float v = o[t][r] * inv;
      ctx[((size_t)(b * TT + (q0 + qi))) * DIM + h * DH + (t << 4) + ln] = (bf16_t)v;
    }
  }
}

// ---------------- host side ----------------

extern "C" void kernel_launch(void* const* d_in, const int* in_sizes, int n_in,
                              void* d_out, int out_size, void* d_ws, size_t ws_size,
                              hipStream_t stream) {
  (void)in_sizes; (void)n_in; (void)out_size; (void)ws_size;
  const float* x   = (const float*)d_in[0];
  const float* pos = (const float*)d_in[1];
  const unsigned char* mask = (const unsigned char*)d_in[2];
  const float* Wq = (const float*)d_in[3];
  const float* bq = (const float*)d_in[4];
  const float* Wk = (const float*)d_in[5];
  const float* bk = (const float*)d_in[6];
  const float* Wv = (const float*)d_in[7];
  const float* bv = (const float*)d_in[8];
  const float* Wp = (const float*)d_in[9];
  const float* Wo = (const float*)d_in[10];
  const float* bo = (const float*)d_in[11];
  const float* bu = (const float*)d_in[12];
  const float* bvb = (const float*)d_in[13];
  float* out = (float*)d_out;

  char* ws = (char*)d_ws;
  size_t off = 0;
  auto alloc = [&](size_t bytes) -> char* {
    char* p = ws + off;
    off += (bytes + 255) & ~(size_t)255;
    return p;
  };
  bf16_t* x_bf   = (bf16_t*)alloc((size_t)BB * TT * DIM * 2);
  bf16_t* pos_bf = (bf16_t*)alloc((size_t)TT * DIM * 2);
  bf16_t* wq_bf  = (bf16_t*)alloc((size_t)DIM * DIM * 2);
  bf16_t* wk_bf  = (bf16_t*)alloc((size_t)DIM * DIM * 2);
  bf16_t* wv_bf  = (bf16_t*)alloc((size_t)DIM * DIM * 2);
  bf16_t* wp_bf  = (bf16_t*)alloc((size_t)DIM * DIM * 2);
  bf16_t* wo_bf  = (bf16_t*)alloc((size_t)DIM * DIM * 2);
  bf16_t* qu_buf = (bf16_t*)alloc((size_t)BB * NH * TT * DH * 2);
  bf16_t* qv_buf = (bf16_t*)alloc((size_t)BB * NH * TT * DH * 2);
  bf16_t* k_buf  = (bf16_t*)alloc((size_t)BB * NH * TT * DH * 2);
  bf16_t* vT_buf = (bf16_t*)alloc((size_t)BB * NH * DH * TT * 2);
  bf16_t* p_buf  = (bf16_t*)alloc((size_t)NH * TT * DH * 2);
  bf16_t* ctx_bf = (bf16_t*)alloc((size_t)BB * TT * DIM * 2);

  // 1) fp32 -> bf16 conversions
  auto cvt = [&](const float* s, bf16_t* d, int n) {
    cvt_f32_bf16<<<(n + 255) / 256, 256, 0, stream>>>(s, d, n);
  };
  cvt(x,   x_bf,   BB * TT * DIM);
  cvt(pos, pos_bf, TT * DIM);
  cvt(Wq,  wq_bf,  DIM * DIM);
  cvt(Wk,  wk_bf,  DIM * DIM);
  cvt(Wv,  wv_bf,  DIM * DIM);
  cvt(Wp,  wp_bf,  DIM * DIM);
  cvt(Wo,  wo_bf,  DIM * DIM);

  // 2) projections (WMMA GEMMs), 16x64 per wave
  dim3 blk(32);
  dim3 gP((BB * TT) / 16, DIM / 64);
  proj_kernel<<<gP, blk, 0, stream>>>(x_bf, wq_bf, bq, bu, bvb, qu_buf, qv_buf, nullptr, 0);
  proj_kernel<<<gP, blk, 0, stream>>>(x_bf, wk_bf, bk, nullptr, nullptr, k_buf, nullptr, nullptr, 1);
  proj_kernel<<<gP, blk, 0, stream>>>(x_bf, wv_bf, bv, nullptr, nullptr, vT_buf, nullptr, nullptr, 2);
  dim3 gPp(TT / 16, DIM / 64);
  proj_kernel<<<gPp, blk, 0, stream>>>(pos_bf, wp_bf, nullptr, nullptr, nullptr, p_buf, nullptr, nullptr, 3);

  // 3) flash attention with exact relative shift
  dim3 gA(TT / 16, NH, BB);
  attn_kernel<<<gA, blk, 0, stream>>>(qu_buf, qv_buf, k_buf, vT_buf, p_buf, mask, ctx_bf);

  // 4) output projection -> fp32
  proj_kernel<<<gP, blk, 0, stream>>>(ctx_bf, wo_bf, bo, nullptr, nullptr, nullptr, nullptr, out, 4);
}
